// TtLlamaAttention_20272245637546
// MI455X (gfx1250) — matrix operand order repaired and checked by reading, not compiled, not run on hardware
//
#include <hip/hip_runtime.h>
#include <cmath>

typedef __bf16 bf16;
typedef __attribute__((ext_vector_type(4)))  bf16  v4bf;
typedef __attribute__((ext_vector_type(8)))  bf16  v8bf;
typedef __attribute__((ext_vector_type(16))) bf16  v16bf;
typedef __attribute__((ext_vector_type(8)))  float v8f;

#define HIDDEN 4096
#define SEQ    2048
#define NHEADS 32
#define HDIM   128

// Scheduling-group control: 0x008 = MFMA/WMMA class, 0x100 = DS read.
#if __has_builtin(__builtin_amdgcn_sched_group_barrier)
#define SGB(mask, size) __builtin_amdgcn_sched_group_barrier(mask, size, 0)
#else
#define SGB(mask, size)
#endif

// Build a 16-element bf16 fragment from two 16-byte LDS/global chunks.
__device__ __forceinline__ v16bf mk16(const bf16* lo, const bf16* hi) {
  v8bf l = *reinterpret_cast<const v8bf*>(lo);
  v8bf h = *reinterpret_cast<const v8bf*>(hi);
  return __builtin_shufflevector(l, h, 0,1,2,3,4,5,6,7,8,9,10,11,12,13,14,15);
}

__device__ __forceinline__ v8f wmma_bf16(v16bf a, v16bf b, v8f c) {
  return __builtin_amdgcn_wmma_f32_16x16x32_bf16(
      /*neg_a=*/false, a, /*neg_b=*/false, b,
      /*c_mod=*/(short)0, c, /*reuse_a=*/false, /*reuse_b=*/false);
}

// Async global->LDS copies (ASYNCcnt-tracked). INST_OFFSET applies to both
// the LDS and global addresses, so one base pair + immediate offsets moves a
// contiguous run.
__device__ __forceinline__ void async_copy_64B(void* lds_dst, const void* gsrc) {
  unsigned dst = (unsigned)(size_t)lds_dst;  // low 32 bits = LDS offset
  asm volatile(
      "global_load_async_to_lds_b128 %0, %1, off\n\t"
      "global_load_async_to_lds_b128 %0, %1, off offset:16\n\t"
      "global_load_async_to_lds_b128 %0, %1, off offset:32\n\t"
      "global_load_async_to_lds_b128 %0, %1, off offset:48"
      :: "v"(dst), "v"(gsrc) : "memory");
}
__device__ __forceinline__ void async_copy_32B(void* lds_dst, const void* gsrc) {
  unsigned dst = (unsigned)(size_t)lds_dst;
  asm volatile(
      "global_load_async_to_lds_b128 %0, %1, off\n\t"
      "global_load_async_to_lds_b128 %0, %1, off offset:16"
      :: "v"(dst), "v"(gsrc) : "memory");
}
__device__ __forceinline__ void wait_async0() {
  asm volatile("s_wait_asynccnt 0x0" ::: "memory");
}

// ---------------------------------------------------------------- convert
__global__ void f32_to_bf16_kernel(const float* __restrict__ x,
                                   bf16* __restrict__ y, int n4) {
  int i = blockIdx.x * blockDim.x + threadIdx.x;
  if (i >= n4) return;
  float4 v = reinterpret_cast<const float4*>(x)[i];
  v4bf o;
  o[0] = (bf16)v.x; o[1] = (bf16)v.y; o[2] = (bf16)v.z; o[3] = (bf16)v.w;
  reinterpret_cast<v4bf*>(y)[i] = o;
}

// ---------------------------------------------------------------- GEMM
// C[m,n] = sum_k A[m,k] * W[n,k]. A bf16 (async->LDS, ping-pong), W f32
// converted through regs. Block tile 128x128, K-step 32; waves 4(M) x 2(N).
template <typename OutT>
__global__ __launch_bounds__(256)
void gemm_xwT(const bf16* __restrict__ A, const float* __restrict__ W,
              OutT* __restrict__ C) {
  __shared__ bf16 As[2][128][40];
  __shared__ bf16 Bs[2][128][40];

  const int t    = threadIdx.x;
  const int wave = t >> 5, lane = t & 31;
  const int hl   = lane >> 4, l16 = lane & 15;
  const int wm   = wave & 3,  wn  = wave >> 2;
  const int m0   = blockIdx.y * 128, n0 = blockIdx.x * 128;

  const v8f vz = {0.f,0.f,0.f,0.f,0.f,0.f,0.f,0.f};
  v8f acc[2][4];
  for (int i = 0; i < 2; ++i)
    for (int j = 0; j < 4; ++j) acc[i][j] = vz;

  const int lrow  = t >> 1;         // 2 threads per tile row
  const int lhalf = (t & 1) * 16;   // 16 elements each

  float4 wf[4];
  const float* wrow = W + (size_t)(n0 + lrow) * HIDDEN + lhalf;
  const bf16*  arow = A + (size_t)(m0 + lrow) * HIDDEN + lhalf;

  auto loadW = [&](int k0) {
    const float4* src = reinterpret_cast<const float4*>(wrow + k0);
    wf[0] = src[0]; wf[1] = src[1]; wf[2] = src[2]; wf[3] = src[3];
  };
  auto storeW = [&](int buf) {
    v8bf c0, c1;
    c0[0]=(bf16)wf[0].x; c0[1]=(bf16)wf[0].y; c0[2]=(bf16)wf[0].z; c0[3]=(bf16)wf[0].w;
    c0[4]=(bf16)wf[1].x; c0[5]=(bf16)wf[1].y; c0[6]=(bf16)wf[1].z; c0[7]=(bf16)wf[1].w;
    c1[0]=(bf16)wf[2].x; c1[1]=(bf16)wf[2].y; c1[2]=(bf16)wf[2].z; c1[3]=(bf16)wf[2].w;
    c1[4]=(bf16)wf[3].x; c1[5]=(bf16)wf[3].y; c1[6]=(bf16)wf[3].z; c1[7]=(bf16)wf[3].w;
    *reinterpret_cast<v8bf*>(&Bs[buf][lrow][lhalf])     = c0;
    *reinterpret_cast<v8bf*>(&Bs[buf][lrow][lhalf + 8]) = c1;
  };

  // Prologue: tile 0 in flight.
  async_copy_32B(&As[0][lrow][lhalf], arow);
  loadW(0);

  const int NK = HIDDEN / 32;
  for (int it = 0; it < NK; ++it) {
    const int buf = it & 1;
    storeW(buf);                 // publish W tile it (regs -> LDS)
    wait_async0();               // A tile it resident
    __syncthreads();             // tile it visible to all waves
    if (it + 1 < NK) {           // start tile it+1 while computing it
      async_copy_32B(&As[buf ^ 1][lrow][lhalf], arow + (it + 1) * 32);
      loadW((it + 1) * 32);
      if (it + 2 < NK)
        __builtin_prefetch(wrow + (it + 2) * 32, 0, 0);
    }

    v16bf afr[2], bfr[4];
#pragma unroll
    for (int mi = 0; mi < 2; ++mi) {
      const bf16* p = &As[buf][wm * 32 + mi * 16 + l16][hl * 8];
      afr[mi] = mk16(p, p + 16);
    }
#pragma unroll
    for (int ni = 0; ni < 4; ++ni) {
      const bf16* p = &Bs[buf][wn * 64 + ni * 16 + l16][hl * 8];
      bfr[ni] = mk16(p, p + 16);
    }
#pragma unroll
    for (int mi = 0; mi < 2; ++mi)
#pragma unroll
      for (int ni = 0; ni < 4; ++ni)
        acc[mi][ni] = wmma_bf16(afr[mi], bfr[ni], acc[mi][ni]);

    // Pipeline shape: 4 DS-reads head, then (1 WMMA, 1 DS-read) x 8.
    SGB(0x100, 4);
#pragma unroll
    for (int i = 0; i < 8; ++i) { SGB(0x008, 1); SGB(0x100, 1); }
  }

  for (int mi = 0; mi < 2; ++mi)
    for (int ni = 0; ni < 4; ++ni)
      for (int r = 0; r < 8; ++r) {
        int grow = m0 + wm * 32 + mi * 16 + r + 8 * hl;
        int gcol = n0 + wn * 64 + ni * 16 + l16;
        C[(size_t)grow * HIDDEN + gcol] = (OutT)acc[mi][ni][r];
      }
}

// ---------------------------------------------------------------- RoPE
__global__ void rope_kernel(bf16* __restrict__ Q, bf16* __restrict__ K,
                            const int* __restrict__ pos_ids) {
  int idx = blockIdx.x * blockDim.x + threadIdx.x;   // s*32*64 + h*64 + j
  int j = idx & 63;
  int h = (idx >> 6) & 31;
  int s = idx >> 11;
  if (s >= SEQ) return;
  float p   = (float)pos_ids[s];
  float inv = powf(10000.0f, -(float)(2 * j) / 128.0f);
  float fr  = p * inv;
  float c = cosf(fr), sn = sinf(fr);
  size_t base = (size_t)s * HIDDEN + h * HDIM + j;
  float q1 = (float)Q[base], q2 = (float)Q[base + 64];
  float k1 = (float)K[base], k2 = (float)K[base + 64];
  const float sc = 0.08838834764831845f;  // 1/sqrt(128)
  Q[base]      = (bf16)((q1 * c - q2 * sn) * sc);
  Q[base + 64] = (bf16)((q2 * c + q1 * sn) * sc);
  K[base]      = (bf16)(k1 * c - k2 * sn);
  K[base + 64] = (bf16)(k2 * c + k1 * sn);
}

// ---------------------------------------------------------------- flash attention
// Grid: (SEQ/128 query tiles, NHEADS). 8 waves; wave w owns 16 query rows.
// Double-buffered 64-key tiles: K via async->LDS, V via regs + LDS transpose.
__global__ __launch_bounds__(256)
void flash_attn(const bf16* __restrict__ Q, const bf16* __restrict__ K,
                const bf16* __restrict__ V, bf16* __restrict__ O) {
  __shared__ bf16 Ks_[2][64][136];   // [buf][key][dim]
  __shared__ bf16 Vt_[2][128][72];   // [buf][dim][key] transposed
  __shared__ bf16 Ps_[8][16][72];    // per-wave P staging [wave][row][key]

  const int t    = threadIdx.x;
  const int wave = t >> 5, lane = t & 31;
  const int hl   = lane >> 4, l16 = lane & 15;
  const int mt   = blockIdx.x, h = blockIdx.y;
  const int q0w  = mt * 128 + wave * 16;

  const int lrow = t >> 2;          // 0..63: one key row per 4 threads
  const int lseg = (t & 3) * 32;    // 32 dims each
  const int ktmax = 2 * mt + 1;     // causal: keys up to mt*128+127

  const bf16* krow = K + (size_t)lrow * HIDDEN + h * HDIM + lseg;
  const bf16* vrow = V + (size_t)lrow * HIDDEN + h * HDIM + lseg;

  bf16 vtmp[32];
  auto loadV = [&](int kt) {
    const uint4* src = reinterpret_cast<const uint4*>(vrow + (size_t)kt * 64 * HIDDEN);
    reinterpret_cast<uint4*>(vtmp)[0] = src[0];
    reinterpret_cast<uint4*>(vtmp)[1] = src[1];
    reinterpret_cast<uint4*>(vtmp)[2] = src[2];
    reinterpret_cast<uint4*>(vtmp)[3] = src[3];
  };

  // Prologue: tile 0 in flight; Q fragments overlap the transfer.
  async_copy_64B(&Ks_[0][lrow][lseg], krow);
  loadV(0);

  v16bf qfr[4];
  {
    const bf16* qrowp = Q + (size_t)(q0w + l16) * HIDDEN + h * HDIM;
#pragma unroll
    for (int kf = 0; kf < 4; ++kf) {
      const bf16* p = qrowp + kf * 32 + hl * 8;
      qfr[kf] = mk16(p, p + 16);
    }
  }

  const v8f vz = {0.f,0.f,0.f,0.f,0.f,0.f,0.f,0.f};
  v8f o[8];
  for (int i = 0; i < 8; ++i) o[i] = vz;
  float m_i[8], l_i[8];
  for (int r = 0; r < 8; ++r) { m_i[r] = -3.0e38f; l_i[r] = 0.f; }

  for (int kt = 0; kt <= ktmax; ++kt) {
    const int buf = kt & 1;
    const int k0  = kt * 64;

    // Publish V tile kt (regs -> transposed LDS).
#pragma unroll
    for (int d = 0; d < 32; ++d) Vt_[buf][lseg + d][lrow] = vtmp[d];
    wait_async0();               // K tile kt resident in LDS
    __syncthreads();             // tile kt visible to all waves
    if (kt < ktmax) {            // start tile kt+1 while computing kt
      async_copy_64B(&Ks_[buf ^ 1][lrow][lseg],
                     krow + (size_t)(kt + 1) * 64 * HIDDEN);
      loadV(kt + 1);
    }

    // S = Q K^T (16 queries x 64 keys per wave, 16 WMMAs / 32 DS-reads).
    v8f s[4];
    for (int nt = 0; nt < 4; ++nt) s[nt] = vz;
    {
#pragma unroll
      for (int idx = 0; idx < 16; ++idx) {   // idx = kf*4 + nt
        const bf16* p = &Ks_[buf][(idx & 3) * 16 + l16][(idx >> 2) * 32 + hl * 8];
        v16bf b = mk16(p, p + 16);
        s[idx & 3] = wmma_bf16(qfr[idx >> 2], b, s[idx & 3]);
      }
      // Pipeline shape: 4 reads head, 14 x (1 WMMA, 2 reads), 2 tail WMMAs.
      SGB(0x100, 4);
#pragma unroll
      for (int i = 0; i < 14; ++i) { SGB(0x008, 1); SGB(0x100, 2); }
      SGB(0x008, 1); SGB(0x008, 1);
    }

    // Causal mask on diagonal-overlapping tiles.
    if (k0 + 63 > q0w) {
      for (int nt = 0; nt < 4; ++nt)
        for (int r = 0; r < 8; ++r) {
          int key = k0 + nt * 16 + l16;
          int qr  = q0w + r + 8 * hl;
          if (key > qr) s[nt][r] = -1.0e9f;
        }
    }

    // Online softmax. Row M = r + 8*hl lives across the 16 lanes of a half.
    for (int r = 0; r < 8; ++r) {
      float rmax = s[0][r];
      for (int nt = 1; nt < 4; ++nt) rmax = fmaxf(rmax, s[nt][r]);
      for (int off = 1; off < 16; off <<= 1)
        rmax = fmaxf(rmax, __shfl_xor(rmax, off, 16));
      float m_new = fmaxf(m_i[r], rmax);
      float alpha = __expf(m_i[r] - m_new);
      float rsum = 0.f;
      for (int nt = 0; nt < 4; ++nt) {
        float pv = __expf(s[nt][r] - m_new);
        s[nt][r] = pv;
        rsum += pv;
      }
      for (int off = 1; off < 16; off <<= 1)
        rsum += __shfl_xor(rsum, off, 16);
      l_i[r] = l_i[r] * alpha + rsum;
      m_i[r] = m_new;
      for (int ni = 0; ni < 8; ++ni) o[ni][r] *= alpha;
    }

    // Re-layout P (C-layout) into A-fragments via per-wave LDS staging.
    for (int nt = 0; nt < 4; ++nt)
      for (int r = 0; r < 8; ++r)
        Ps_[wave][r + 8 * hl][nt * 16 + l16] = (bf16)s[nt][r];
    // Same-wave LDS ops are processed in order; no barrier needed.

    // O += P * V (16 WMMAs / 36 DS-reads incl. the two A fragments).
    {
      const bf16* pp0 = &Ps_[wave][l16][hl * 8];
      const bf16* pp1 = &Ps_[wave][l16][32 + hl * 8];
      v16bf a0 = mk16(pp0, pp0 + 16);
      v16bf a1 = mk16(pp1, pp1 + 16);
#pragma unroll
      for (int idx = 0; idx < 16; ++idx) {   // idx = kf*8 + ni
        const bf16* p = &Vt_[buf][(idx & 7) * 16 + l16][(idx >> 3) * 32 + hl * 8];
        v16bf b = mk16(p, p + 16);
        o[idx & 7] = wmma_bf16((idx >> 3) ? a1 : a0, b, o[idx & 7]);
      }
      // Pipeline shape: 8 reads head, 14 x (1 WMMA, 2 reads), 2 tail WMMAs.
      SGB(0x100, 8);
#pragma unroll
      for (int i = 0; i < 14; ++i) { SGB(0x008, 1); SGB(0x100, 2); }
      SGB(0x008, 1); SGB(0x008, 1);
    }
  }

  for (int ni = 0; ni < 8; ++ni)
    for (int r = 0; r < 8; ++r) {
      int row = q0w + r + 8 * hl;
      int col = h * HDIM + ni * 16 + l16;
      O[(size_t)row * HIDDEN + col] = (bf16)(o[ni][r] / l_i[r]);
    }
}

// ---------------------------------------------------------------- launch
extern "C" void kernel_launch(void* const* d_in, const int* in_sizes, int n_in,
                              void* d_out, int out_size, void* d_ws, size_t ws_size,
                              hipStream_t stream) {
  const float* hidden = (const float*)d_in[0];
  // d_in[1] = attention_mask: exact causal mask, applied analytically.
  const float* wq = (const float*)d_in[2];
  const float* wk = (const float*)d_in[3];
  const float* wv = (const float*)d_in[4];
  const float* wo = (const float*)d_in[5];
  const int*  pos = (const int*)d_in[6];

  const size_t XB = (size_t)SEQ * HIDDEN;  // elements per activation buffer
  bf16* Xb = (bf16*)d_ws;
  bf16* Qb = Xb + XB;
  bf16* Kb = Qb + XB;
  bf16* Vb = Kb + XB;
  bf16* Ab = Vb + XB;

  f32_to_bf16_kernel<<<(int)((XB / 4 + 255) / 256), 256, 0, stream>>>(
      hidden, Xb, (int)(XB / 4));

  dim3 ggrid(HIDDEN / 128, SEQ / 128);
  gemm_xwT<bf16><<<ggrid, 256, 0, stream>>>(Xb, wq, Qb);
  gemm_xwT<bf16><<<ggrid, 256, 0, stream>>>(Xb, wk, Kb);
  gemm_xwT<bf16><<<ggrid, 256, 0, stream>>>(Xb, wv, Vb);

  rope_kernel<<<(SEQ * NHEADS * 64) / 256, 256, 0, stream>>>(Qb, Kb, pos);

  flash_attn<<<dim3(SEQ / 128, NHEADS), 256, 0, stream>>>(Qb, Kb, Vb, Ab);

  gemm_xwT<float><<<ggrid, 256, 0, stream>>>(Ab, wo, (float*)d_out);
}
